// CausalSubgraphNet_61117384622467
// MI455X (gfx1250) — matrix operand order repaired and checked by reading, not compiled, not run on hardware
//
#include <hip/hip_runtime.h>

// ---------------------------------------------------------------------------
// CausalSubgraphNet for MI455X (gfx1250): bf16 WMMA GEMMs (b128 fragment
// loads, pre-swizzled weight fragments, software-pipelined K loop), fused
// edge MLP, atomic-scatter GCN aggregation, bitonic argsort, split outputs.
// ---------------------------------------------------------------------------

#define Nn    50000
#define Ee    800000
#define HIDc  128
#define OUTc  128
#define NCONF 160000          // int((1-0.8)*E)
#define NCAUS (Ee - NCONF)
#define PADN  (1 << 20)       // bitonic pad (>= Ee)

typedef __attribute__((ext_vector_type(16))) __bf16 v16bf;
typedef __attribute__((ext_vector_type(8)))  float  v8f;

struct U32x8 { uint4 lo, hi; };   // 32B == one v16bf fragment

__device__ __forceinline__ v16bf frag_cast(uint4 lo, uint4 hi) {
  U32x8 t{lo, hi};
  return __builtin_bit_cast(v16bf, t);
}

// float -> bf16 bits (round to nearest even)
__device__ __forceinline__ unsigned short f2bu(float f) {
  union { float f; unsigned u; } x; x.f = f;
  unsigned r = x.u + 0x7FFFu + ((x.u >> 16) & 1u);
  return (unsigned short)(r >> 16);
}
__device__ __forceinline__ __bf16 f2b(float f) {
  unsigned short hs = f2bu(f);
  __bf16 b; __builtin_memcpy(&b, &hs, 2); return b;
}

// ---------------------------------------------------------------------------
// Utility kernels
// ---------------------------------------------------------------------------
__global__ void zero_u32(unsigned* p, size_t n) {
  size_t i = (size_t)blockIdx.x * blockDim.x + threadIdx.x;
  if (i < n) p[i] = 0u;
}

// Pack B[K x Nc] (f32, row-major) into WMMA-fragment-ordered bf16 pairs:
// Bp[(kt*(Nc/16) + nt)*256 + lane*8 + v] = {B[k, n], B[k+1, n]}
// with k = kt*32 + (v<4 ? (lane>>4)*8 + 2v : 16 + (lane>>4)*8 + 2(v-4)), n = nt*16 + (lane&15).
// Each lane's per-(kt,nt) fragment is then 32 contiguous bytes.
__global__ void pack_b(const float* __restrict__ B, unsigned* __restrict__ Bp,
                       int K, int Nc) {
  size_t i = (size_t)blockIdx.x * blockDim.x + threadIdx.x;
  size_t total = (size_t)K * Nc / 2;
  if (i >= total) return;
  int v    = (int)(i & 7);
  int lane = (int)((i >> 3) & 31);
  size_t rest = i >> 8;
  int ntiles = Nc / 16;
  int nt = (int)(rest % ntiles);
  int kt = (int)(rest / ntiles);
  int h = lane >> 4, m = lane & 15;
  int k = kt * 32 + (v < 4 ? h * 8 + 2 * v : 16 + h * 8 + 2 * (v - 4));
  int n = nt * 16 + m;
  unsigned lo = f2bu(B[(size_t)k * Nc + n]);
  unsigned hi = f2bu(B[(size_t)(k + 1) * Nc + n]);
  Bp[i] = lo | (hi << 16);
}

// degree counts: deg_row for embedding, deg_col for GCN normalization
__global__ void count_deg(const long long* row, const long long* col,
                          int* deg_row, int* deg_col, int E_) {
  int e = blockIdx.x * blockDim.x + threadIdx.x;
  if (e >= E_) return;
  atomicAdd(&deg_row[(int)row[e]], 1);
  atomicAdd(&deg_col[(int)col[e]], 1);
}

// x2 = x + deg_emb[min(deg_row,99)]  (written directly as bf16 GEMM operand)
__global__ void node_prep(const float* x, const float* deg_emb,
                          const int* deg_row, __bf16* x2bf) {
  size_t i = (size_t)blockIdx.x * blockDim.x + threadIdx.x;
  if (i >= (size_t)Nn * HIDc) return;
  int node = (int)(i / HIDc), f = (int)(i % HIDc);
  int d = deg_row[node]; d = d > 99 ? 99 : d;
  x2bf[i] = f2b(x[i] + deg_emb[(size_t)d * HIDc + f]);
}

// dinv = rsqrt(deg_col + 1)  (+1 == self loop)
__global__ void calc_dinv(const int* deg_col, float* dinv) {
  int i = blockIdx.x * blockDim.x + threadIdx.x;
  if (i < Nn) dinv[i] = rsqrtf((float)deg_col[i] + 1.0f);
}

// ---------------------------------------------------------------------------
// Wave-level bf16 WMMA GEMM: C[M x Nc] = A[M x K] * Bp (fragment-packed).
// grid = (M/16, Nc/64), block = 32. Each wave computes a 16x64 C strip.
// Two-stage software pipeline: fragments for kt+1 are in flight while the
// 4 WMMAs for kt execute, so waits are partial instead of full drains.
// ---------------------------------------------------------------------------
__global__ void gemm_bf16(const __bf16* __restrict__ A,
                          const unsigned* __restrict__ Bp,
                          float* __restrict__ C, int Nc, int K) {
  const int lane = threadIdx.x;
  const int h = lane >> 4, m = lane & 15;
  const int m0 = blockIdx.x * 16;
  const int nt0 = blockIdx.y * 4;
  const int ntiles = Nc / 16;
  const int NK = K >> 5;

  const __bf16* Arow = A + (size_t)(m0 + m) * K + h * 8;

  v8f acc[4] = {};
  uint4 ca0, ca1, cb[8];

  // prologue: stage kt = 0
  ca0 = *(const uint4*)(Arow);
  ca1 = *(const uint4*)(Arow + 16);
#pragma unroll
  for (int t = 0; t < 4; ++t) {
    const uint4* bp = (const uint4*)(Bp + (((size_t)(nt0 + t)) * 32 + lane) * 8);
    cb[2 * t] = bp[0]; cb[2 * t + 1] = bp[1];
  }

  for (int kt = 0; kt < NK - 1; ++kt) {
    // stage kt+1 while kt computes
    uint4 na0 = *(const uint4*)(Arow + (kt + 1) * 32);
    uint4 na1 = *(const uint4*)(Arow + (kt + 1) * 32 + 16);
    uint4 nb[8];
#pragma unroll
    for (int t = 0; t < 4; ++t) {
      const uint4* bp =
          (const uint4*)(Bp + (((size_t)(kt + 1) * ntiles + nt0 + t) * 32 + lane) * 8);
      nb[2 * t] = bp[0]; nb[2 * t + 1] = bp[1];
    }
    v16bf a = frag_cast(ca0, ca1);
#pragma unroll
    for (int t = 0; t < 4; ++t)
      acc[t] = __builtin_amdgcn_wmma_f32_16x16x32_bf16(
          false, a, false, frag_cast(cb[2 * t], cb[2 * t + 1]),
          (short)0, acc[t], false, false);
    ca0 = na0; ca1 = na1;
#pragma unroll
    for (int i = 0; i < 8; ++i) cb[i] = nb[i];
  }
  // epilogue: last kt
  {
    v16bf a = frag_cast(ca0, ca1);
#pragma unroll
    for (int t = 0; t < 4; ++t)
      acc[t] = __builtin_amdgcn_wmma_f32_16x16x32_bf16(
          false, a, false, frag_cast(cb[2 * t], cb[2 * t + 1]),
          (short)0, acc[t], false, false);
  }

#pragma unroll
  for (int t = 0; t < 4; ++t) {
    int n = (nt0 + t) * 16 + m;
#pragma unroll
    for (int v = 0; v < 8; ++v) {
      int r = m0 + h * 8 + v;
      C[(size_t)r * Nc + n] = acc[t][v];
    }
  }
}

// ---------------------------------------------------------------------------
// GCN scatter: out[col] += xw[row] * dinv[row]*dinv[col]   (one wave / edge)
// ---------------------------------------------------------------------------
__global__ void scatter_edges(const float* __restrict__ XW,
                              const long long* __restrict__ row,
                              const long long* __restrict__ col,
                              const float* __restrict__ dinv,
                              float* __restrict__ Out, int E_, int F) {
  int warp = (blockIdx.x * blockDim.x + threadIdx.x) >> 5;
  int lane = threadIdx.x & 31;
  if (warp >= E_) return;
  int r = (int)row[warp], c = (int)col[warp];
  float nrm = dinv[r] * dinv[c];
  const float* src = XW + (size_t)r * F;
  float* dst = Out + (size_t)c * F;
  for (int j = lane; j < F; j += 32) atomicAdd(&dst[j], src[j] * nrm);
}

// self loop + bias (+ optional time_emb) (+ optional relu), dual f32/bf16 out
__global__ void selfloop_finish(const float* __restrict__ XW,
                                const float* __restrict__ dinv,
                                const float* __restrict__ bias,
                                const float* __restrict__ time_emb, // nullable
                                const int* __restrict__ tptr,       // nullable
                                float* __restrict__ Out,
                                __bf16* __restrict__ OutBf,
                                int F, int do_relu) {
  size_t i = (size_t)blockIdx.x * blockDim.x + threadIdx.x;
  if (i >= (size_t)Nn * F) return;
  int node = (int)(i / F), f = (int)(i % F);
  float di = dinv[node];
  float v = Out[i] + XW[i] * di * di + bias[f];
  if (time_emb) v += time_emb[(size_t)tptr[0] * F + f];
  if (do_relu) v = fmaxf(v, 0.0f);
  Out[i] = v;
  OutBf[i] = f2b(v);
}

// ---------------------------------------------------------------------------
// Fused edge MLP: per block = 16 edges (one wave).
// A = [h[row] | h[col]] (16 x 256 bf16 staged in LDS via b128 copies).
// Sweep 512 hidden cols in 8 groups of 4 N-tiles with a two-stage pipeline on
// the K loop; bias+relu in-register, dot with mW2 into LDS accumulator ->
// sigmoid score. The 800000x512 hid tensor never touches memory.
// ---------------------------------------------------------------------------
__global__ void edge_mlp(const __bf16* __restrict__ Hbf,
                         const long long* __restrict__ row,
                         const long long* __restrict__ col,
                         const unsigned* __restrict__ mW1p,  // packed fragments
                         const float* __restrict__ mb1,
                         const float* __restrict__ mW2,
                         const float* __restrict__ mb2,
                         float* __restrict__ score) {
  __shared__ __attribute__((aligned(16))) __bf16 Atile[16 * 256];
  __shared__ float sacc[16];
  const int lane = threadIdx.x;
  const int e0 = blockIdx.x * 16;

  // stage A tile: 512 chunks of 8 bf16 (16B each); chunks never straddle
  // the row/col half boundary (128 is a multiple of 8).
  for (int idx = lane; idx < 512; idx += 32) {
    int er = idx >> 5;            // edge within tile
    int f0 = (idx & 31) * 8;      // concat feature base
    int node = (f0 < OUTc) ? (int)row[e0 + er] : (int)col[e0 + er];
    *(uint4*)&Atile[er * 256 + f0] =
        *(const uint4*)&Hbf[(size_t)node * OUTc + (f0 & (OUTc - 1))];
  }
  if (lane < 16) sacc[lane] = 0.0f;
  __syncthreads();

  const int h = lane >> 4, m = lane & 15;
  const __bf16* Ar0 = &Atile[m * 256 + h * 8];

  for (int ntg = 0; ntg < 8; ++ntg) {
    v8f acc[4] = {};
    uint4 ca0, ca1, cb[8];
    // prologue: kt = 0
    ca0 = *(const uint4*)(Ar0);
    ca1 = *(const uint4*)(Ar0 + 16);
#pragma unroll
    for (int t = 0; t < 4; ++t) {
      const uint4* bp =
          (const uint4*)(mW1p + (((size_t)(ntg * 4 + t)) * 32 + lane) * 8);
      cb[2 * t] = bp[0]; cb[2 * t + 1] = bp[1];
    }
#pragma unroll
    for (int kt = 0; kt < 7; ++kt) {
      uint4 na0 = *(const uint4*)(Ar0 + (kt + 1) * 32);
      uint4 na1 = *(const uint4*)(Ar0 + (kt + 1) * 32 + 16);
      uint4 nb[8];
#pragma unroll
      for (int t = 0; t < 4; ++t) {
        const uint4* bp = (const uint4*)(mW1p +
            (((size_t)(kt + 1) * 32 + ntg * 4 + t) * 32 + lane) * 8);
        nb[2 * t] = bp[0]; nb[2 * t + 1] = bp[1];
      }
      v16bf a = frag_cast(ca0, ca1);
#pragma unroll
      for (int t = 0; t < 4; ++t)
        acc[t] = __builtin_amdgcn_wmma_f32_16x16x32_bf16(
            false, a, false, frag_cast(cb[2 * t], cb[2 * t + 1]),
            (short)0, acc[t], false, false);
      ca0 = na0; ca1 = na1;
#pragma unroll
      for (int i = 0; i < 8; ++i) cb[i] = nb[i];
    }
    {
      v16bf a = frag_cast(ca0, ca1);
#pragma unroll
      for (int t = 0; t < 4; ++t)
        acc[t] = __builtin_amdgcn_wmma_f32_16x16x32_bf16(
            false, a, false, frag_cast(cb[2 * t], cb[2 * t + 1]),
            (short)0, acc[t], false, false);
    }

#pragma unroll
    for (int t = 0; t < 4; ++t) {
      int coln = (ntg * 4 + t) * 16 + m;
      float w2 = mW2[coln], bb = mb1[coln];
#pragma unroll
      for (int v = 0; v < 8; ++v) {
        float val = fmaxf(acc[t][v] + bb, 0.0f);   // relu(hid)
        atomicAdd(&sacc[h * 8 + v], val * w2);     // dot with mW2
      }
    }
  }
  __syncthreads();
  if (lane < 16) {
    float s = sacc[lane] + mb2[0];
    score[e0 + lane] = 1.0f / (1.0f + __expf(-s));
  }
}

// ---------------------------------------------------------------------------
// Sort: pad keys with +huge so valid edges occupy the front ascending.
// ---------------------------------------------------------------------------
__global__ void init_sort(float* keys, int* vals) {
  int i = blockIdx.x * blockDim.x + threadIdx.x;
  if (i >= PADN) return;
  keys[i] = 3.0e38f;   // real scores (<=1) written afterwards for i < Ee
  vals[i] = i;
}

__global__ void bitonic_step(float* keys, int* vals, unsigned j, unsigned k) {
  unsigned i = blockIdx.x * blockDim.x + threadIdx.x;
  unsigned ixj = i ^ j;
  if (ixj <= i) return;
  float ki = keys[i], kj = keys[ixj];
  bool up = ((i & k) == 0);
  if ((ki > kj) == up) {
    keys[i] = kj; keys[ixj] = ki;
    int vi = vals[i]; vals[i] = vals[ixj]; vals[ixj] = vi;
  }
}

// ---------------------------------------------------------------------------
// Output split: [causal_ei(2 x NCAUS i64) | conf_ei(2 x NCONF i64) |
//                causal_score(NCAUS f32) | conf_score(NCONF f32)]
// ---------------------------------------------------------------------------
__global__ void write_outputs(const float* __restrict__ keys,
                              const int* __restrict__ vals,
                              const long long* __restrict__ ei,
                              char* __restrict__ out) {
  int t = blockIdx.x * blockDim.x + threadIdx.x;
  if (t >= Ee) return;
  int e = vals[t];
  float s = keys[t];
  long long r = ei[e], c = ei[Ee + e];
  long long* causal_ei = (long long*)out;
  long long* conf_ei   = (long long*)(out + (size_t)2 * NCAUS * 8);
  float* causal_s = (float*)(out + (size_t)2 * NCAUS * 8 + (size_t)2 * NCONF * 8);
  float* conf_s   = causal_s + NCAUS;
  if (t < NCONF) {
    conf_ei[t] = r; conf_ei[NCONF + t] = c; conf_s[t] = s;
  } else {
    int u = t - NCONF;
    causal_ei[u] = r; causal_ei[NCAUS + u] = c; causal_s[u] = s;
  }
}

// ---------------------------------------------------------------------------
// Host driver
// ---------------------------------------------------------------------------
extern "C" void kernel_launch(void* const* d_in, const int* in_sizes, int n_in,
                              void* d_out, int out_size, void* d_ws, size_t ws_size,
                              hipStream_t stream) {
  const float*     x        = (const float*)d_in[0];
  const long long* ei       = (const long long*)d_in[1];   // (2, E) int64
  const int*       tptr     = (const int*)d_in[2];
  const float*     deg_emb  = (const float*)d_in[3];
  const float*     W1       = (const float*)d_in[4];
  const float*     b1       = (const float*)d_in[5];
  const float*     W2       = (const float*)d_in[6];
  const float*     b2       = (const float*)d_in[7];
  const float*     time_emb = (const float*)d_in[8];
  const float*     mW1      = (const float*)d_in[9];
  const float*     mb1      = (const float*)d_in[10];
  const float*     mW2      = (const float*)d_in[11];
  const float*     mb2      = (const float*)d_in[12];
  const long long* row = ei;
  const long long* col = ei + Ee;

  // --- carve workspace (~215 MB) ---
  char* ws = (char*)d_ws;
  size_t off = 0;
  auto carve = [&](size_t bytes) -> char* {
    char* p = ws + off;
    off = (off + bytes + 255) & ~(size_t)255;
    return p;
  };
  int*      deg_row = (int*)carve(sizeof(int) * Nn);
  int*      deg_col = (int*)carve(sizeof(int) * Nn);
  float*    dinv    = (float*)carve(sizeof(float) * Nn);
  __bf16*   x2bf    = (__bf16*)carve(2ull * Nn * HIDc);
  unsigned* W1p     = (unsigned*)carve(2ull * HIDc * 2 * HIDc);       // packed
  unsigned* W2p     = (unsigned*)carve(2ull * 2 * HIDc * OUTc);       // packed
  unsigned* mW1p    = (unsigned*)carve(2ull * 2 * OUTc * 4 * OUTc);   // packed
  float*    xw      = (float*)carve(4ull * Nn * 2 * HIDc);   // x2 @ W1
  float*    h1      = (float*)carve(4ull * Nn * 2 * HIDc);   // conv1 out
  __bf16*   h1bf    = (__bf16*)carve(2ull * Nn * 2 * HIDc);
  float*    xw2     = (float*)carve(4ull * Nn * OUTc);       // h1 @ W2
  float*    h2      = (float*)carve(4ull * Nn * OUTc);       // conv2 out
  __bf16*   h2bf    = (__bf16*)carve(2ull * Nn * OUTc);
  float*    keys    = (float*)carve(4ull * PADN);
  int*      vals    = (int*)carve(4ull * PADN);
  (void)ws_size; (void)in_sizes; (void)n_in; (void)out_size;

  const int TB = 256;
  auto blk = [](size_t n, int t) { return (unsigned)((n + t - 1) / t); };

  // 1) zero accumulators + degree counters
  zero_u32<<<blk(2 * Nn, TB), TB, 0, stream>>>((unsigned*)deg_row, 2 * Nn);
  zero_u32<<<blk((size_t)Nn * 2 * HIDc, TB), TB, 0, stream>>>((unsigned*)h1, (size_t)Nn * 2 * HIDc);
  zero_u32<<<blk((size_t)Nn * OUTc, TB), TB, 0, stream>>>((unsigned*)h2, (size_t)Nn * OUTc);

  // 2) degrees, node prep, normalization
  count_deg<<<blk(Ee, TB), TB, 0, stream>>>(row, col, deg_row, deg_col, Ee);
  node_prep<<<blk((size_t)Nn * HIDc, TB), TB, 0, stream>>>(x, deg_emb, deg_row, x2bf);
  calc_dinv<<<blk(Nn, TB), TB, 0, stream>>>(deg_col, dinv);

  // 3) pack weights into WMMA fragment order (bf16 pairs)
  pack_b<<<blk((size_t)HIDc * 2 * HIDc / 2, TB), TB, 0, stream>>>(W1, W1p, HIDc, 2 * HIDc);
  pack_b<<<blk((size_t)2 * HIDc * OUTc / 2, TB), TB, 0, stream>>>(W2, W2p, 2 * HIDc, OUTc);
  pack_b<<<blk((size_t)2 * OUTc * 4 * OUTc / 2, TB), TB, 0, stream>>>(mW1, mW1p, 2 * OUTc, 4 * OUTc);

  // 4) conv1: xw = x2 @ W1 ; scatter ; self-loop + b1 + relu
  gemm_bf16<<<dim3(Nn / 16, (2 * HIDc) / 64), 32, 0, stream>>>(x2bf, W1p, xw, 2 * HIDc, HIDc);
  scatter_edges<<<blk((size_t)Ee * 32, TB), TB, 0, stream>>>(xw, row, col, dinv, h1, Ee, 2 * HIDc);
  selfloop_finish<<<blk((size_t)Nn * 2 * HIDc, TB), TB, 0, stream>>>(
      xw, dinv, b1, nullptr, nullptr, h1, h1bf, 2 * HIDc, 1);

  // 5) conv2: xw2 = h1 @ W2 ; scatter ; self-loop + b2 + time_emb[t]
  gemm_bf16<<<dim3(Nn / 16, OUTc / 64), 32, 0, stream>>>(h1bf, W2p, xw2, OUTc, 2 * HIDc);
  scatter_edges<<<blk((size_t)Ee * 32, TB), TB, 0, stream>>>(xw2, row, col, dinv, h2, Ee, OUTc);
  selfloop_finish<<<blk((size_t)Nn * OUTc, TB), TB, 0, stream>>>(
      xw2, dinv, b2, time_emb, tptr, h2, h2bf, OUTc, 0);

  // 6) fused edge MLP -> scores (keys)
  init_sort<<<blk(PADN, TB), TB, 0, stream>>>(keys, vals);
  edge_mlp<<<Ee / 16, 32, 0, stream>>>(h2bf, row, col, mW1p, mb1, mW2, mb2, keys);

  // 7) bitonic argsort ascending over padded 2^20
  for (unsigned k = 2; k <= PADN; k <<= 1)
    for (unsigned j = k >> 1; j > 0; j >>= 1)
      bitonic_step<<<PADN / TB, TB, 0, stream>>>(keys, vals, j, k);

  // 8) split into conf / causal outputs
  write_outputs<<<blk(Ee, TB), TB, 0, stream>>>(keys, vals, ei, (char*)d_out);
}
